// MentalPongIN_44968307589269
// MI455X (gfx1250) — matrix operand order repaired and checked by reading, not compiled, not run on hardware
//
#include <hip/hip_runtime.h>

typedef _Float16 v8h  __attribute__((ext_vector_type(8)));
typedef _Float16 v16h __attribute__((ext_vector_type(16)));
typedef float    v8f  __attribute__((ext_vector_type(8)));

#define TSTEPS 256
#define BATCH  8192
#define BB     32      // batch rows per block
#define MROWS  64      // relation rows per block (BB*2)
#define S_H    168     // K-stride (halves) for h buffer (bank-conflict-free for b128)
#define S_RI   72      // K-stride for rel_in
#define S_EF   24      // stride for effect buffer
#define S_W1   64      // K-stride for staged W1^T

// A-fragment: 16x32 f16 (ISA 7.12.2): lane = row (mod 16); lanes 0-15 hold
// K {kb..kb+7, kb+16..kb+23}, lanes 16-31 hold K {kb+8..kb+15, kb+24..kb+31}
__device__ __forceinline__ v16h load_a(const _Float16* p, int stride, int m, int kb, int half) {
  const v8h* q0 = (const v8h*)(p + m * stride + kb + half * 8);
  const v8h* q1 = (const v8h*)(p + m * stride + kb + 16 + half * 8);
  v16h r;
  ((v8h*)&r)[0] = *q0;
  ((v8h*)&r)[1] = *q1;
  return r;
}

// B-fragment: 32x16 f16 from W^T rows (row = output col n, contiguous in K).
// lane = column (mod 16); lanes 0-15 hold K {kb..kb+7, kb+16..kb+23}, lanes 16-31 hold K {kb+8..kb+15, kb+24..kb+31}
__device__ __forceinline__ v16h load_b(const _Float16* w, int stride, int n, int kb, int half) {
  const v8h* q0 = (const v8h*)(w + n * stride + kb + half * 8);
  const v8h* q1 = (const v8h*)(w + n * stride + kb + 16 + half * 8);
  v16h r;
  ((v8h*)&r)[0] = *q0;
  ((v8h*)&r)[1] = *q1;
  return r;
}

__device__ __forceinline__ v8f wmma16(v16h a, v16h b, v8f c) {
  return __builtin_amdgcn_wmma_f32_16x16x32_f16(false, a, false, b, (short)0, c, false, false);
}

// Hidden layer: h = relu(h @ W + b), in-place on hb. Weights live in VGPRs (w),
// two independent M-blocks per wave (r0, r1) -> two independent WMMA chains per B-frag.
__device__ __forceinline__ void gemm_hid2(_Float16* hb, const v16h (&w)[5][5], const float (&bv)[5],
                                          int r0, int r1, int nB, int lrow, int half) {
  v16h a0[5], a1[5];
  #pragma unroll
  for (int ks = 0; ks < 5; ++ks) {
    a0[ks] = load_a(hb, S_H, r0 + lrow, ks * 32, half);
    a1[ks] = load_a(hb, S_H, r1 + lrow, ks * 32, half);
  }
  __syncthreads();            // everyone's A in registers before overwriting hb
  #pragma unroll
  for (int nt = 0; nt < 5; ++nt) {
    v8f c0 = {0.f,0.f,0.f,0.f,0.f,0.f,0.f,0.f};
    v8f c1 = {0.f,0.f,0.f,0.f,0.f,0.f,0.f,0.f};
    #pragma unroll
    for (int ks = 0; ks < 5; ++ks) {
      c0 = wmma16(a0[ks], w[nt][ks], c0);
      c1 = wmma16(a1[ks], w[nt][ks], c1);
    }
    const int col = nB + nt * 16 + lrow;
    const float bias = bv[nt];
    #pragma unroll
    for (int v = 0; v < 8; ++v) {
      float x0 = fmaxf(c0[v] + bias, 0.f);
      float x1 = fmaxf(c1[v] + bias, 0.f);
      hb[(r0 + v + 8 * half) * S_H + col] = (_Float16)x0;
      hb[(r1 + v + 8 * half) * S_H + col] = (_Float16)x1;
    }
  }
  __syncthreads();
}

__global__ __launch_bounds__(128, 1) void in_recurrent_kernel(
    const float* __restrict__ x,
    const float* __restrict__ W1, const float* __restrict__ b1,
    const float* __restrict__ W2, const float* __restrict__ b2,
    const float* __restrict__ W3, const float* __restrict__ b3,
    const float* __restrict__ W4, const float* __restrict__ b4,
    const float* __restrict__ Wo, const float* __restrict__ bo,
    float* __restrict__ outs, float* __restrict__ effs)
{
  extern __shared__ char smem[];
  _Float16* ST = (_Float16*)smem;             // staging region: 160*S_H halves (reused below)
  _Float16* ri = ST;                          // [64][S_RI]  rel_in   (4608 halves)
  _Float16* hb = ri + MROWS * S_RI;           // [64][S_H]   hidden   (10752 halves)
  _Float16* ef = hb + MROWS * S_H;            // [64][S_EF]  effects  (1536 halves)  (total 16896 < 26880)
  float* fb  = (float*)(ST + 160 * S_H);      // persistent float region
  float* b1s = fb;          // 160
  float* b2s = fb + 160;
  float* b3s = fb + 320;
  float* b4s = fb + 480;    // 16
  float* bos = fb + 496;    // 8
  float* Wos = fb + 504;    // 140

  const int tid   = threadIdx.x;
  const int bBase = blockIdx.x * BB;
  const int wave  = tid >> 5, lane = tid & 31;
  const int mg    = wave & 1;          // m-group: rows [mg*32, mg*32+32)
  const int nh    = wave >> 1;         // N-half: cols [nh*80, nh*80+80)
  const int nB    = nh * 80;
  const int lrow  = lane & 15;
  const int half  = lane >> 4;
  const int r0    = mg * 32;           // first 16-row m-block
  const int r1    = mg * 32 + 16;     // second 16-row m-block

  // ---- persistent small constants ----
  for (int e = tid; e < 160; e += 128) {
    b1s[e] = (e < 150) ? b1[e] : 0.f;
    b2s[e] = (e < 150) ? b2[e] : 0.f;
    b3s[e] = (e < 150) ? b3[e] : 0.f;
  }
  if (tid < 16) b4s[tid] = (tid < 10) ? b4[tid] : 0.f;
  if (tid < 8)  bos[tid] = (tid < 7)  ? bo[tid] : 0.f;
  for (int e = tid; e < 140; e += 128) Wos[e] = Wo[e];

  // ---- stage each weight through LDS once, grab fragments into VGPRs ----
  v16h w1[5][2], w2[5][5], w3[5][5], w4[5];

  for (int e = tid; e < 160 * S_W1; e += 128) {   // W1^T [160][64], K 35->64 zero-pad
    int n = e >> 6, k = e & 63;
    ST[n * S_W1 + k] = (k < 35 && n < 150) ? (_Float16)W1[k * 150 + n] : (_Float16)0.f;
  }
  __syncthreads();
  #pragma unroll
  for (int nt = 0; nt < 5; ++nt)
    #pragma unroll
    for (int ks = 0; ks < 2; ++ks)
      w1[nt][ks] = load_b(ST, S_W1, nB + nt * 16 + lrow, ks * 32, half);
  __syncthreads();

  for (int e = tid; e < 160 * 160; e += 128) {    // W2^T [160][S_H]
    int n = e / 160, k = e - n * 160;
    ST[n * S_H + k] = (k < 150 && n < 150) ? (_Float16)W2[k * 150 + n] : (_Float16)0.f;
  }
  __syncthreads();
  #pragma unroll
  for (int nt = 0; nt < 5; ++nt)
    #pragma unroll
    for (int ks = 0; ks < 5; ++ks)
      w2[nt][ks] = load_b(ST, S_H, nB + nt * 16 + lrow, ks * 32, half);
  __syncthreads();

  for (int e = tid; e < 160 * 160; e += 128) {    // W3^T
    int n = e / 160, k = e - n * 160;
    ST[n * S_H + k] = (k < 150 && n < 150) ? (_Float16)W3[k * 150 + n] : (_Float16)0.f;
  }
  __syncthreads();
  #pragma unroll
  for (int nt = 0; nt < 5; ++nt)
    #pragma unroll
    for (int ks = 0; ks < 5; ++ks)
      w3[nt][ks] = load_b(ST, S_H, nB + nt * 16 + lrow, ks * 32, half);
  __syncthreads();

  for (int e = tid; e < 16 * 160; e += 128) {     // W4^T [16][S_H], N 10->16
    int n = e / 160, k = e - n * 160;
    ST[n * S_H + k] = (k < 150 && n < 10) ? (_Float16)W4[k * 10 + n] : (_Float16)0.f;
  }
  __syncthreads();
  if (nh == 0) {                                  // wave-uniform
    #pragma unroll
    for (int ks = 0; ks < 5; ++ks)
      w4[ks] = load_b(ST, S_H, lrow, ks * 32, half);
  }
  __syncthreads();

  // per-lane bias registers
  float b1v[5], b2v[5], b3v[5];
  #pragma unroll
  for (int nt = 0; nt < 5; ++nt) {
    b1v[nt] = b1s[nB + nt * 16 + lrow];
    b2v[nt] = b2s[nB + nt * 16 + lrow];
    b3v[nt] = b3s[nB + nt * 16 + lrow];
  }
  const float b4v = b4s[lrow];

  // zero rel_in (pads stay 0 forever) and effect carry
  for (int e = tid; e < MROWS * S_RI; e += 128) ri[e] = (_Float16)0.f;
  for (int e = tid; e < MROWS * S_EF; e += 128) ef[e] = (_Float16)0.f;
  __syncthreads();

  #pragma unroll 1
  for (int t = 0; t < TSTEPS; ++t) {
    // ---- build rel_in: row = 2*b + r; sender = obj r, receiver = obj (1-r) ----
    for (int e = tid; e < BB * 14; e += 128) {
      int b = e / 14, rem = e - b * 14, j = rem / 7, k = rem - j * 7;
      float v = x[((size_t)(bBase + b) * TSTEPS + t) * 14 + rem];
      _Float16 hv = (_Float16)v;
      ri[(2 * b + j) * S_RI + k] = hv;
      ri[(2 * b + (1 - j)) * S_RI + 17 + k] = hv;
    }
    for (int e = tid; e < MROWS * 10; e += 128) {
      int r = e / 10, d = e - r * 10;
      _Float16 ev = ef[r * S_EF + d];
      ri[r * S_RI + 7 + d] = ev;
      ri[(r ^ 1) * S_RI + 24 + d] = ev;
    }
    __syncthreads();

    // ---- layer 1: rel_in[64x64] @ W1 ----
    {
      v16h a0[2], a1[2];
      #pragma unroll
      for (int ks = 0; ks < 2; ++ks) {
        a0[ks] = load_a(ri, S_RI, r0 + lrow, ks * 32, half);
        a1[ks] = load_a(ri, S_RI, r1 + lrow, ks * 32, half);
      }
      #pragma unroll
      for (int nt = 0; nt < 5; ++nt) {
        v8f c0 = {0.f,0.f,0.f,0.f,0.f,0.f,0.f,0.f};
        v8f c1 = {0.f,0.f,0.f,0.f,0.f,0.f,0.f,0.f};
        #pragma unroll
        for (int ks = 0; ks < 2; ++ks) {
          c0 = wmma16(a0[ks], w1[nt][ks], c0);
          c1 = wmma16(a1[ks], w1[nt][ks], c1);
        }
        const int col = nB + nt * 16 + lrow;
        const float bias = b1v[nt];
        #pragma unroll
        for (int v = 0; v < 8; ++v) {
          float x0 = fmaxf(c0[v] + bias, 0.f);
          float x1 = fmaxf(c1[v] + bias, 0.f);
          hb[(r0 + v + 8 * half) * S_H + col] = (_Float16)x0;
          hb[(r1 + v + 8 * half) * S_H + col] = (_Float16)x1;
        }
      }
    }
    __syncthreads();

    // ---- layers 2 & 3 (weights in VGPRs) ----
    gemm_hid2(hb, w2, b2v, r0, r1, nB, lrow, half);
    gemm_hid2(hb, w3, b3v, r0, r1, nB, lrow, half);

    // ---- layer 4: effects = h @ W4 + b4; swap rows -> effect_recv ----
    if (nh == 0) {   // wave-uniform branch, EXEC all-ones inside
      v16h a0[5], a1[5];
      #pragma unroll
      for (int ks = 0; ks < 5; ++ks) {
        a0[ks] = load_a(hb, S_H, r0 + lrow, ks * 32, half);
        a1[ks] = load_a(hb, S_H, r1 + lrow, ks * 32, half);
      }
      v8f c0 = {0.f,0.f,0.f,0.f,0.f,0.f,0.f,0.f};
      v8f c1 = {0.f,0.f,0.f,0.f,0.f,0.f,0.f,0.f};
      #pragma unroll
      for (int ks = 0; ks < 5; ++ks) {
        c0 = wmma16(a0[ks], w4[ks], c0);
        c1 = wmma16(a1[ks], w4[ks], c1);
      }
      if (lrow < 10) {
        #pragma unroll
        for (int v = 0; v < 8; ++v) {
          int m0 = r0 + v + 8 * half;
          int m1 = r1 + v + 8 * half;
          float e0 = c0[v] + b4v;
          float e1 = c1[v] + b4v;
          ef[(m0 ^ 1) * S_EF + lrow] = (_Float16)e0;   // effect_recv[2b+o] = effects[2b+(1-o)]
          ef[(m1 ^ 1) * S_EF + lrow] = (_Float16)e1;
          effs[(((size_t)(bBase + (m0 >> 1)) * TSTEPS + t) * 2 + (1 - (m0 & 1))) * 10 + lrow] = e0;
          effs[(((size_t)(bBase + (m1 >> 1)) * TSTEPS + t) * 2 + (1 - (m1 & 1))) * 10 + lrow] = e1;
        }
      }
    }
    __syncthreads();

    // ---- output head: outs[b,t] = effect_recv.flat(20) @ Wo + bo ----
    if (tid < BB) {
      float acc[7];
      #pragma unroll
      for (int j = 0; j < 7; ++j) acc[j] = bos[j];
      for (int o2 = 0; o2 < 2; ++o2)
        for (int d = 0; d < 10; ++d) {
          float e2 = (float)ef[(2 * tid + o2) * S_EF + d];
          #pragma unroll
          for (int j = 0; j < 7; ++j) acc[j] += e2 * Wos[(o2 * 10 + d) * 7 + j];
        }
      size_t ob = ((size_t)(bBase + tid) * TSTEPS + t) * 7;
      #pragma unroll
      for (int j = 0; j < 7; ++j) outs[ob + j] = acc[j];
    }
    // ef: read here & in next build (reads only); next write is behind 3+ barriers
  }
}

extern "C" void kernel_launch(void* const* d_in, const int* in_sizes, int n_in,
                              void* d_out, int out_size, void* d_ws, size_t ws_size,
                              hipStream_t stream) {
  const float* x  = (const float*)d_in[0];
  const float* W1 = (const float*)d_in[1];
  const float* b1 = (const float*)d_in[2];
  const float* W2 = (const float*)d_in[3];
  const float* b2 = (const float*)d_in[4];
  const float* W3 = (const float*)d_in[5];
  const float* b3 = (const float*)d_in[6];
  const float* W4 = (const float*)d_in[7];
  const float* b4 = (const float*)d_in[8];
  const float* Wo = (const float*)d_in[9];
  const float* bo = (const float*)d_in[10];
  float* outs = (float*)d_out;                          // [B,T,7]
  float* effs = outs + (size_t)BATCH * TSTEPS * 7;      // [B,T,2,10]

  size_t shmem = (size_t)160 * S_H * sizeof(_Float16)   // staging region (reused as ri/hb/ef)
               + (size_t)(504 + 140) * sizeof(float);   // biases + head weights  => ~56.3 KB

  in_recurrent_kernel<<<dim3(BATCH / BB), dim3(128), shmem, stream>>>(
      x, W1, b1, W2, b2, W3, b3, W4, b4, Wo, bo, outs, effs);
}
